// _SpikingClassifier_80908593922077
// MI455X (gfx1250) — matrix operand order repaired and checked by reading, not compiled, not run on hardware
//
#include <hip/hip_runtime.h>

// ---------------------------------------------------------------------------
// Spiking LIF classifier for gfx1250 (MI455X), wave32 + WMMA bf16 + TDM.
//   B=64, T=512, D=256, H=512, C=100
// Pipeline:
//   k_xA / k_WinB / k_WrecB : fp32 -> bf16 swizzle into WMMA fragment layouts
//   k_init                  : zero spike ping buffer + global barrier counter
//   k_ingemm                : cur_in[t] = x[:,t,:] @ W_in^T + b_in  (WMMA bf16)
//   k_scan                  : persistent 512-step LIF scan, 8 WGs, W_rec in LDS,
//                             cur_in double-buffered into LDS by the Tensor
//                             Data Mover (tensor_load_to_lds + s_wait_tensorcnt),
//                             v/rate in registers, global gen-counter barrier
//   k_out                   : logits = rate @ W_out^T + b_out
// ---------------------------------------------------------------------------

typedef __attribute__((ext_vector_type(16))) __bf16 v16bf;
typedef __attribute__((ext_vector_type(8)))  float  v8f;
typedef __attribute__((ext_vector_type(4)))  unsigned v4u;
typedef __attribute__((ext_vector_type(8)))  int     v8i;
typedef __attribute__((ext_vector_type(4)))  int     v4i;

#define B_  64
#define T_  512
#define D_  256
#define H_  512
#define C_  100
#define NWG 8

__device__ __forceinline__ unsigned short f2bf(float f) {
    unsigned u = __float_as_uint(f);
    unsigned r = u + 0x7FFFu + ((u >> 16) & 1u);   // round-to-nearest-even
    return (unsigned short)(r >> 16);
}

// Issue a TDM 2D tile load: 4 rows x 4096 B, global row stride 32 KB -> LDS.
// D# group0: count=1 | lds_addr | global_addr[56:0] | type=2
// D# group1: data_size=4B, tensor_dim0=1024, tensor_dim1=4, tile 1024x4,
//            tensor_dim0_stride=8192 elements.
__device__ __forceinline__ void tdm_load_cur(unsigned long long ga, unsigned lds_off) {
    v4u g0;
    g0[0] = 1u;                                                   // count = 1
    g0[1] = lds_off;                                              // LDS byte addr
    g0[2] = (unsigned)ga;                                         // global[31:0]
    g0[3] = ((unsigned)(ga >> 32) & 0x01FFFFFFu) | 0x80000000u;   // global[56:32] | type=2
    v8i g1;
    g1[0] = 0x00020000;              // workgroup_mask=0, data_size=2 (4 bytes)
    g1[1] = (int)(1024u << 16);      // tensor_dim0[15:0] = 1024
    g1[2] = (int)(4u << 16);         // tensor_dim0[31:16]=0 | tensor_dim1 = 4
    g1[3] = (int)(1024u << 16);      // tensor_dim1 hi = 0 | tile_dim0 = 1024
    g1[4] = 4;                       // tile_dim1 = 4, tile_dim2 = 0
    g1[5] = 8192;                    // tensor_dim0_stride = 8192 elements (32 KB)
    g1[6] = 0;
    g1[7] = 0;
    v4i z4 = {};
#if defined(__clang_major__) && (__clang_major__ >= 23)
    v8i z8 = {};
    __builtin_amdgcn_tensor_load_to_lds(g0, g1, z4, z4, z8, 0);
#else
    __builtin_amdgcn_tensor_load_to_lds(g0, g1, z4, z4, 0);
#endif
}

// ---- A-fragment layout (bf16 16x32 MxK): lane L, elem e -> (m,k) -------------
//   m = L&15 ; hs = L>>4 ; v = e>>1 ; lo = e&1
//   k = (v<4 ? 2v : 16+2(v-4)) + 8*hs + lo
// ---- B-fragment layout (bf16 32x16 KxN): n = L&15 ; k = (e<8 ? e : e+8) + 8*(L>>4) --------
// ---- C/D fragment (f32 16x16):  n = L&15 ; m = r + 8*(L>>4) ------------------

// x (B,T,D) fp32 -> xA[t][mt(4)][kt(8)][lane(32)][e(16)] bf16  (A-frags per t)
__global__ void k_xA(const float* __restrict__ x, unsigned short* __restrict__ xA) {
    unsigned id = blockIdx.x * blockDim.x + threadIdx.x;   // 8,388,608 total
    unsigned e    = id & 15u;
    unsigned lane = (id >> 4) & 31u;
    unsigned kt   = (id >> 9) & 7u;
    unsigned mt   = (id >> 12) & 3u;
    unsigned t    = id >> 14;
    unsigned m  = lane & 15u, hs = lane >> 4;
    unsigned v  = e >> 1,     lo = e & 1u;
    unsigned k  = ((v < 4u) ? (2u * v) : (16u + 2u * (v - 4u))) + 8u * hs + lo;
    unsigned b  = mt * 16u + m;
    unsigned d  = kt * 32u + k;
    xA[id] = f2bf(x[((size_t)b * T_ + t) * D_ + d]);
}

// W_in (H,D) fp32 -> WinB[kt(8)][nt(32)][lane][e] bf16  (B = W_in^T fragments)
__global__ void k_WinB(const float* __restrict__ Win, unsigned short* __restrict__ WinB) {
    unsigned id = blockIdx.x * blockDim.x + threadIdx.x;   // 131,072 total
    unsigned e    = id & 15u;
    unsigned lane = (id >> 4) & 31u;
    unsigned nt   = (id >> 9) & 31u;
    unsigned kt   = id >> 14;                               // 0..7
    unsigned h = nt * 16u + (lane & 15u);
    unsigned d = kt * 32u + ((e < 8u) ? e : (e + 8u)) + 8u * (lane >> 4);
    WinB[id] = f2bf(Win[(size_t)h * D_ + d]);
}

// W_rec (H,H) fp32 -> WrecB[kt(16)][nt(32)][lane][e] bf16 (B = W_rec^T frags)
__global__ void k_WrecB(const float* __restrict__ Wrec, unsigned short* __restrict__ WrecB) {
    unsigned id = blockIdx.x * blockDim.x + threadIdx.x;   // 262,144 total
    unsigned e    = id & 15u;
    unsigned lane = (id >> 4) & 31u;
    unsigned nt   = (id >> 9) & 31u;
    unsigned kt   = id >> 14;                               // 0..15
    unsigned h = nt * 16u + (lane & 15u);
    unsigned k = kt * 32u + ((e < 8u) ? e : (e + 8u)) + 8u * (lane >> 4);
    WrecB[id] = f2bf(Wrec[(size_t)h * H_ + k]);
}

__global__ void k_init(unsigned* __restrict__ spike0_u32, unsigned* __restrict__ cnt) {
    unsigned id = blockIdx.x * blockDim.x + threadIdx.x;
    if (id < 16384u) spike0_u32[id] = 0u;                   // 32768 bf16 zeros
    if (id == 0u)    *cnt = 0u;
}

// cur_in: (t, B, H) in D-fragment layout [t][mt(4)][nt(32)][lane(32)][r(8)] f32
__global__ void __launch_bounds__(256)
k_ingemm(const unsigned short* __restrict__ xA,
         const unsigned short* __restrict__ WinB,
         const float* __restrict__ b_in,
         float* __restrict__ cur) {
    unsigned t    = blockIdx.x >> 2;
    unsigned g    = blockIdx.x & 3u;        // group of 8 N-tiles
    unsigned lane = threadIdx.x & 31u;
    unsigned wid  = threadIdx.x >> 5;
    unsigned mt   = wid >> 1;
    unsigned half = wid & 1u;

    v8f acc[4] = {};
    for (unsigned kt = 0; kt < 8u; ++kt) {
        v16bf a = *(const v16bf*)(xA + ((((size_t)t * 4u + mt) * 8u + kt) * 32u + lane) * 16u);
#pragma unroll
        for (int j = 0; j < 4; ++j) {
            unsigned nt = g * 8u + half * 4u + (unsigned)j;
            v16bf bb = *(const v16bf*)(WinB + (((size_t)kt * 32u + nt) * 32u + lane) * 16u);
            acc[j] = __builtin_amdgcn_wmma_f32_16x16x32_bf16(
                false, a, false, bb, (short)0, acc[j], false, false);
        }
    }
#pragma unroll
    for (int j = 0; j < 4; ++j) {
        unsigned nt = g * 8u + half * 4u + (unsigned)j;
        float bias = b_in[nt * 16u + (lane & 15u)];
#pragma unroll
        for (int r = 0; r < 8; ++r) acc[j][r] += bias;
        *(v8f*)(cur + ((((size_t)t * 4u + mt) * 32u + nt) * 32u + lane) * 8u) = acc[j];
    }
}

// Persistent LIF scan. 8 WGs x 256 threads; WG `slice` owns H columns
// [slice*64, slice*64+64). Spike buffers are global, A-fragment layout
// [mt(4)][kt(16)][lane(32)][e(16)] bf16, ping-ponged each step. The per-step
// feed-forward current tile (16 KB) is DMA'd into LDS one step ahead by the
// Tensor Data Mover (wave 0 issues; TENSORcnt is per-wave and in-order, so
// waiting <=1 after issuing t+1 guarantees step t's tile is resident).
__global__ void __launch_bounds__(256, 1)
k_scan(const unsigned short* __restrict__ WrecB,
       const float* __restrict__ cur,
       unsigned short* __restrict__ sbuf0,
       unsigned short* __restrict__ sbuf1,
       float* __restrict__ rate,
       unsigned* __restrict__ cnt) {
    __shared__ unsigned short wlds[16 * 4 * 32 * 16];       // 64 KB W_rec frags
    __shared__ float curlds[2][4096];                       // 2 x 16 KB cur tiles
    const unsigned slice = blockIdx.x;                      // 0..7
    const unsigned lane  = threadIdx.x & 31u;
    const unsigned wid   = threadIdx.x >> 5;
    const unsigned mt    = wid & 3u;
    const unsigned np    = wid >> 2;
    const unsigned ntgb  = slice * 4u;

    // Stage this slice's W_rec^T B-fragments into LDS once (reused 512x).
    for (unsigned fi = wid; fi < 64u; fi += 8u) {
        unsigned kt = fi >> 2, ntl = fi & 3u;
        *(v16bf*)&wlds[((kt * 4u + ntl) * 32u + lane) * 16u] =
            *(const v16bf*)(WrecB + (((size_t)kt * 32u + (ntgb + ntl)) * 32u + lane) * 16u);
    }

    // TDM bookkeeping: flat shared pointer low 32 bits == LDS byte offset.
    const unsigned long long cur_base =
        (unsigned long long)(size_t)cur + (unsigned long long)slice * 4096ull;
    const unsigned ldsb0 = (unsigned)(size_t)&curlds[0][0];
    const unsigned ldsb1 = (unsigned)(size_t)&curlds[1][0];
    if (wid == 0u) {
        tdm_load_cur(cur_base, ldsb0);                      // prefetch t = 0
    }
    __syncthreads();

    v8f v[2] = {}, rsum[2] = {};
    const float alpha = 0.9048374180359595f;                // exp(-1/10)

    for (unsigned t = 0; t < T_; ++t) {
        const unsigned short* sp = (t & 1u) ? sbuf1 : sbuf0;
        unsigned short*       sn = (t & 1u) ? sbuf0 : sbuf1;

        // Issue TDM for t+1, then wait until step t's tile is in LDS.
        if (wid == 0u) {
            if (t + 1u < T_) {
                tdm_load_cur(cur_base + (unsigned long long)(t + 1u) * 131072ull,
                             ((t + 1u) & 1u) ? ldsb1 : ldsb0);
                __builtin_amdgcn_s_wait_tensorcnt(1);
            } else {
                __builtin_amdgcn_s_wait_tensorcnt(0);
            }
        }
        __syncthreads();

        // feed-forward current from LDS (D-fragment layout)
        const float* curt = &curlds[t & 1u][0];
        v8f acc[2];
#pragma unroll
        for (int j = 0; j < 2; ++j) {
            unsigned ntl = np * 2u + (unsigned)j;
            acc[j] = *(const v8f*)(curt + ((mt * 4u + ntl) * 32u + lane) * 8u);
        }

        // recurrent current: s_prev @ W_rec^T over full K = H = 512
        for (unsigned kt = 0; kt < 16u; ++kt) {
            v16bf a = *(const v16bf*)(sp + (((size_t)mt * 16u + kt) * 32u + lane) * 16u);
#pragma unroll
            for (int j = 0; j < 2; ++j) {
                unsigned ntl = np * 2u + (unsigned)j;
                v16bf bb = *(const v16bf*)&wlds[((kt * 4u + ntl) * 32u + lane) * 16u];
                acc[j] = __builtin_amdgcn_wmma_f32_16x16x32_bf16(
                    false, a, false, bb, (short)0, acc[j], false, false);
            }
        }

        // LIF update + scatter spikes into A-fragment layout of next buffer
#pragma unroll
        for (int j = 0; j < 2; ++j) {
            unsigned ntl   = np * 2u + (unsigned)j;
            unsigned n     = lane & 15u;
            unsigned kappa = (ntl & 1u) * 16u + n;          // k within 32-block
            unsigned hsa   = (kappa >> 3) & 1u;
            unsigned k2    = kappa - 8u * hsa;
            unsigned ea    = (k2 < 8u) ? k2 : (k2 - 8u);
            unsigned kta   = slice * 2u + (ntl >> 1);
            unsigned hsD   = lane >> 4;
#pragma unroll
            for (int r = 0; r < 8; ++r) {
                float vv   = alpha * v[j][r] + acc[j][r];
                bool  fire = (vv - 0.3f) > 0.0f;
                v[j][r]    = fire ? 0.0f : vv;              // hard reset
                rsum[j][r] += fire ? 1.0f : 0.0f;
                unsigned ma    = (unsigned)r + 8u * hsD;
                unsigned lanea = ma + 16u * hsa;
                sn[(((size_t)mt * 16u + kta) * 32u + lanea) * 16u + ea] =
                    fire ? (unsigned short)0x3F80u : (unsigned short)0u;  // bf16 1/0
            }
        }

        // device-wide barrier (monotonic generation counter, reset per launch)
        __threadfence();
        __syncthreads();
        if (threadIdx.x == 0) {
            atomicAdd(cnt, 1u);
            unsigned target = (unsigned)NWG * (t + 1u);
            volatile unsigned* vc = cnt;
            while (*vc < target) __builtin_amdgcn_s_sleep(2);
        }
        __syncthreads();
        __threadfence();
    }

    // temporal mean firing rate, natural (B,H) layout for the readout GEMM
#pragma unroll
    for (int j = 0; j < 2; ++j) {
        unsigned ntl = np * 2u + (unsigned)j;
        unsigned h   = slice * 64u + ntl * 16u + (lane & 15u);
        unsigned hsD = lane >> 4;
#pragma unroll
        for (int r = 0; r < 8; ++r) {
            unsigned b = mt * 16u + (unsigned)r + 8u * hsD;
            rate[(size_t)b * H_ + h] = rsum[j][r] * (1.0f / 512.0f);
        }
    }
}

// logits = rate @ W_out^T + b_out  (tiny: 64x100x512, plain VALU)
__global__ void k_out(const float* __restrict__ rate,
                      const float* __restrict__ Wout,
                      const float* __restrict__ b_out,
                      float* __restrict__ out) {
    unsigned id = blockIdx.x * blockDim.x + threadIdx.x;
    if (id >= B_ * C_) return;
    unsigned b = id / C_, c = id % C_;
    const float4* r4 = (const float4*)(rate + (size_t)b * H_);
    const float4* w4 = (const float4*)(Wout + (size_t)c * H_);
    float acc = 0.0f;
    for (int i = 0; i < H_ / 4; ++i) {
        float4 rv = r4[i], wv = w4[i];
        acc += rv.x * wv.x + rv.y * wv.y + rv.z * wv.z + rv.w * wv.w;
    }
    out[id] = acc + b_out[c];
}

extern "C" void kernel_launch(void* const* d_in, const int* in_sizes, int n_in,
                              void* d_out, int out_size, void* d_ws, size_t ws_size,
                              hipStream_t stream) {
    (void)in_sizes; (void)n_in; (void)out_size; (void)ws_size;
    const float* x     = (const float*)d_in[0];
    const float* W_in  = (const float*)d_in[1];
    const float* b_in  = (const float*)d_in[2];
    const float* W_rec = (const float*)d_in[3];
    const float* W_out = (const float*)d_in[4];
    const float* b_out = (const float*)d_in[5];
    float* out = (float*)d_out;

    char* ws = (char*)d_ws;
    size_t off = 0;
    auto alloc = [&](size_t bytes) {
        char* p = ws + off;
        off = (off + bytes + 255u) & ~(size_t)255u;
        return p;
    };
    unsigned short* xA    = (unsigned short*)alloc(8388608ull * 2);   // 16 MB
    unsigned short* WinB  = (unsigned short*)alloc(131072ull * 2);    // 256 KB
    unsigned short* WrecB = (unsigned short*)alloc(262144ull * 2);    // 512 KB
    float*          cur   = (float*)alloc((size_t)T_ * B_ * H_ * 4);  // 64 MB
    unsigned short* s0b   = (unsigned short*)alloc(32768ull * 2);     // 64 KB
    unsigned short* s1b   = (unsigned short*)alloc(32768ull * 2);     // 64 KB
    float*          rate  = (float*)alloc((size_t)B_ * H_ * 4);       // 128 KB
    unsigned*       cnt   = (unsigned*)alloc(256);

    hipLaunchKernelGGL(k_xA,     dim3(32768), dim3(256), 0, stream, x, xA);
    hipLaunchKernelGGL(k_WinB,   dim3(512),   dim3(256), 0, stream, W_in, WinB);
    hipLaunchKernelGGL(k_WrecB,  dim3(1024),  dim3(256), 0, stream, W_rec, WrecB);
    hipLaunchKernelGGL(k_init,   dim3(65),    dim3(256), 0, stream, (unsigned*)s0b, cnt);
    hipLaunchKernelGGL(k_ingemm, dim3(2048),  dim3(256), 0, stream, xA, WinB, b_in, cur);
    hipLaunchKernelGGL(k_scan,   dim3(NWG),   dim3(256), 0, stream, WrecB, cur, s0b, s1b, rate, cnt);
    hipLaunchKernelGGL(k_out,    dim3(25),    dim3(256), 0, stream, rate, W_out, b_out, out);
}
